// GraphMannLayer_12799002542472
// MI455X (gfx1250) — matrix operand order repaired and checked by reading, not compiled, not run on hardware
//
#include <hip/hip_runtime.h>
#include <hip/hip_bf16.h>

// Problem constants
#define Bx 8
#define Tt 64
#define Ff 128
#define Nn 2048
#define Dd 64
#define Uu 256
#define Hh 2          // R + W
#define PPARAMS 262   // H*(D+3) + W*2*D = 134 + 128

typedef __attribute__((ext_vector_type(16))) __bf16          v16bf;
typedef __attribute__((ext_vector_type(8)))  float           v8f;
typedef __attribute__((ext_vector_type(8)))  unsigned short  ushort8;
typedef __attribute__((ext_vector_type(16))) unsigned short  ushort16;

__device__ __forceinline__ float sigmoidf_(float x) { return 1.0f / (1.0f + __expf(-x)); }
__device__ __forceinline__ float softplusf_(float x) { return (x > 20.0f) ? x : log1pf(__expf(x)); }
__device__ __forceinline__ unsigned short f32_to_bf16(float f) {
    unsigned int u = __float_as_uint(f);
    unsigned int lsb = (u >> 16) & 1u;
    u += 0x7fffu + lsb;                       // round-to-nearest-even
    return (unsigned short)(u >> 16);
}

// ---------------------------------------------------------------------------
// Setup 1: column-normalize adjacency, store bf16 TRANSPOSED:
//   adjT[m][n] = adj[n][m] / (sum_n adj[n][m] + 1e-8)
// so each WMMA B-fragment lane reads 16 consecutive K values (one 32B load).
// One thread per column m -> fully independent.
// ---------------------------------------------------------------------------
__global__ __launch_bounds__(256) void prep_adj_kernel(const float* __restrict__ adj,
                                                       unsigned short* __restrict__ adjT) {
    int m = blockIdx.x * 256 + threadIdx.x;   // 0..2047 (output column of the GEMM)
    float s = 0.0f;
    for (int n = 0; n < Nn; ++n) s += adj[(size_t)n * Nn + m];
    float inv = 1.0f / (s + 1e-8f);
    for (int n = 0; n < Nn; ++n)
        adjT[(size_t)m * Nn + n] = f32_to_bf16(adj[(size_t)n * Nn + m] * inv);
}

// ---------------------------------------------------------------------------
// Setup 2: initialize all recurrent state (ws is poisoned; must init per call)
// ---------------------------------------------------------------------------
__global__ __launch_bounds__(256) void init_state_kernel(const float* __restrict__ init_memory,
                                                         float* __restrict__ mem,
                                                         float* __restrict__ prev_w,
                                                         float* __restrict__ hbuf,
                                                         float* __restrict__ cbuf,
                                                         float* __restrict__ rv) {
    size_t idx = (size_t)blockIdx.x * 256 + threadIdx.x;
    const size_t memtot = (size_t)Bx * Nn * Dd;
    if (idx < memtot)          mem[idx]    = init_memory[idx % ((size_t)Nn * Dd)];
    if (idx < Bx * Hh * Nn)    prev_w[idx] = 1.0f / (float)Nn;
    if (idx < Bx * Uu)         { hbuf[idx] = 0.0f; cbuf[idx] = 0.0f; }
    if (idx < Bx * Dd)         rv[idx] = 0.0f;
}

// ---------------------------------------------------------------------------
// Per-step kernel A: LSTM cell + parameter head + addressing parameters.
// One workgroup, 1024 threads (32 waves). Everything staged in LDS.
// ---------------------------------------------------------------------------
__global__ __launch_bounds__(1024) void cell_kernel(const float* __restrict__ x,
                                                    const float* __restrict__ Wx,
                                                    const float* __restrict__ Wh,
                                                    const float* __restrict__ bvec,
                                                    const float* __restrict__ Wp,
                                                    const float* __restrict__ bp,
                                                    const float* __restrict__ rv,
                                                    float* __restrict__ hbuf,
                                                    float* __restrict__ cbuf,
                                                    float* __restrict__ khat,
                                                    float* __restrict__ betav,
                                                    float* __restrict__ gv,
                                                    float* __restrict__ gammav,
                                                    float* __restrict__ erasev,
                                                    float* __restrict__ addv,
                                                    float* __restrict__ out,
                                                    int t) {
    __shared__ float ci[Bx][Ff + Dd];     // 8 x 192
    __shared__ float hl[Bx][Uu];          // 8 x 256
    __shared__ float zl[Bx][4 * Uu];      // 8 x 1024
    __shared__ float pl[Bx][PPARAMS];     // 8 x 262
    const int tid = threadIdx.x;

    for (int idx = tid; idx < Bx * (Ff + Dd); idx += 1024) {
        int b = idx / (Ff + Dd), i = idx % (Ff + Dd);
        ci[b][i] = (i < Ff) ? x[((size_t)b * Tt + t) * Ff + i] : rv[b * Dd + (i - Ff)];
    }
    for (int idx = tid; idx < Bx * Uu; idx += 1024)
        hl[idx >> 8][idx & 255] = hbuf[idx];
    __syncthreads();

    // z = ci @ Wx + h @ Wh + b   (each thread owns one output column j, all 8 batches)
    {
        const int j = tid;
        float acc[Bx];
#pragma unroll
        for (int b = 0; b < Bx; ++b) acc[b] = bvec[j];
        for (int i = 0; i < Ff + Dd; ++i) {
            float w = Wx[i * 1024 + j];
#pragma unroll
            for (int b = 0; b < Bx; ++b) acc[b] += ci[b][i] * w;
        }
        for (int i = 0; i < Uu; ++i) {
            float w = Wh[i * 1024 + j];
#pragma unroll
            for (int b = 0; b < Bx; ++b) acc[b] += hl[b][i] * w;
        }
#pragma unroll
        for (int b = 0; b < Bx; ++b) zl[b][j] = acc[b];
    }
    __syncthreads();

    // LSTM gates
    if (tid < Uu) {
        const int u = tid;
        for (int b = 0; b < Bx; ++b) {
            float zi = zl[b][u], zf = zl[b][Uu + u], zg = zl[b][2 * Uu + u], zo = zl[b][3 * Uu + u];
            float cc = sigmoidf_(zf) * cbuf[b * Uu + u] + sigmoidf_(zi) * tanhf(zg);
            float hh = sigmoidf_(zo) * tanhf(cc);
            cbuf[b * Uu + u] = cc;
            hbuf[b * Uu + u] = hh;
            hl[b][u] = hh;
        }
    }
    __syncthreads();

    // emit h part of the output for this step
    for (int idx = tid; idx < Bx * Uu; idx += 1024) {
        int b = idx >> 8, u = idx & 255;
        out[((size_t)b * Tt + t) * (Uu + Dd) + u] = hl[b][u];
    }

    // p = h @ Wp + bp
    if (tid < PPARAMS) {
        const int q = tid;
        for (int b = 0; b < Bx; ++b) {
            float acc = bp[q];
            for (int i = 0; i < Uu; ++i) acc += hl[b][i] * Wp[i * PPARAMS + q];
            pl[b][q] = acc;
        }
    }
    __syncthreads();

    // addressing parameters per (b, head)
    if (tid < Bx * Hh) {
        int b = tid >> 1, h = tid & 1;
        const float* a = &pl[b][h * (Dd + 3)];
        float sq = 0.0f;
        for (int d = 0; d < Dd; ++d) sq += a[d] * a[d];
        float inv = 1.0f / fmaxf(sqrtf(sq), 1e-12f);
        for (int d = 0; d < Dd; ++d) khat[tid * Dd + d] = a[d] * inv;
        betav[tid]  = softplusf_(a[Dd]);
        gv[tid]     = sigmoidf_(a[Dd + 1]);
        gammav[tid] = softplusf_(a[Dd + 2]) + 1.0f;
    }
    if (tid >= 512 && tid < 512 + Bx * Dd) {
        int idx = tid - 512;
        int b = idx >> 6, d = idx & 63;
        erasev[b * Dd + d] = sigmoidf_(pl[b][Hh * (Dd + 3) + d]);
        addv[b * Dd + d]   = tanhf(pl[b][Hh * (Dd + 3) + Dd + d]);
    }
}

// ---------------------------------------------------------------------------
// Per-step kernel B: cosine similarity + softmax + gate mix  -> bf16 A rows.
// One block per (b,h) row. Also zeroes rowsum for kernel C's atomics.
// ---------------------------------------------------------------------------
__global__ __launch_bounds__(256) void address_kernel(const float* __restrict__ mem,
                                                      const float* __restrict__ khat,
                                                      const float* __restrict__ betav,
                                                      const float* __restrict__ gv,
                                                      const float* __restrict__ prev_w,
                                                      unsigned short* __restrict__ wgA,
                                                      float* __restrict__ rowsum) {
    const int bh = blockIdx.x;          // 0..15
    const int b = bh >> 1;
    const int tid = threadIdx.x;
    __shared__ float red[256];
    __shared__ float kk[Dd];
    if (tid < Dd) kk[tid] = khat[bh * Dd + tid];
    if (tid == 0) rowsum[bh] = 0.0f;
    __syncthreads();

    float vals[8];
    const float* memb = mem + (size_t)b * Nn * Dd;
    const float bet = betav[bh];
#pragma unroll
    for (int i = 0; i < 8; ++i) {
        int n = tid + i * 256;
        const float* mrow = memb + (size_t)n * Dd;
        float dot = 0.0f, sq = 0.0f;
        for (int d = 0; d < Dd; ++d) { float mv = mrow[d]; dot += kk[d] * mv; sq += mv * mv; }
        vals[i] = bet * (dot / fmaxf(sqrtf(sq), 1e-12f));
    }
    float mx = vals[0];
#pragma unroll
    for (int i = 1; i < 8; ++i) mx = fmaxf(mx, vals[i]);
    red[tid] = mx; __syncthreads();
    for (int s = 128; s > 0; s >>= 1) { if (tid < s) red[tid] = fmaxf(red[tid], red[tid + s]); __syncthreads(); }
    const float gmax = red[0];
    __syncthreads();
    float lsum = 0.0f;
#pragma unroll
    for (int i = 0; i < 8; ++i) { vals[i] = __expf(vals[i] - gmax); lsum += vals[i]; }
    red[tid] = lsum; __syncthreads();
    for (int s = 128; s > 0; s >>= 1) { if (tid < s) red[tid] += red[tid + s]; __syncthreads(); }
    const float inv = 1.0f / red[0];
    const float gval = gv[bh];
#pragma unroll
    for (int i = 0; i < 8; ++i) {
        int n = tid + i * 256;
        float wc = vals[i] * inv;
        float wg = gval * wc + (1.0f - gval) * prev_w[bh * Nn + n];
        wgA[bh * Nn + n] = f32_to_bf16(wg);
    }
}

// ---------------------------------------------------------------------------
// Per-step kernel C: dominant GEMM  w_sh = w_g (16 x 2048) @ norm_adj
// (2048 x 2048), bf16 WMMA. One wave per block, one 16-column tile per wave
// (128 blocks). B matrix stored transposed so each lane's fragment is one
// contiguous 32B load. Fused sharpening w_sh^gamma + shuffle-reduced rowsums.
// A/B fragments use contiguous K = 16*half + s (same for both); C/D layout per ISA (wave32).
// ---------------------------------------------------------------------------
__global__ __launch_bounds__(32) void graph_gemm_kernel(const unsigned short* __restrict__ wgA,
                                                        const unsigned short* __restrict__ adjT,
                                                        const float* __restrict__ gammav,
                                                        float* __restrict__ wsharp,
                                                        float* __restrict__ rowsum) {
    const int lane = threadIdx.x;
    const int m16 = lane & 15;
    const int half = lane >> 4;
    const int ncol = blockIdx.x * 16 + m16;                // this wave's output columns

    v8f acc = v8f{};
    const unsigned short* arow = wgA + (size_t)m16 * Nn;   // A row for M = lane%16
    const unsigned short* bcol = adjT + (size_t)ncol * Nn; // B column, contiguous in K

    for (int ks = 0; ks < Nn / 32; ++ks) {
        const int kbase = ks * 32;
        __builtin_prefetch(arow + kbase + 128, 0, 0);      // global_prefetch_b8
        __builtin_prefetch(bcol + kbase + 128, 0, 0);

        // A fragment (16x32 bf16): lane m = lane%16; elems 0..15 -> K = half*16 + j
        // (two aligned 16B loads)
        ushort8 alo = *reinterpret_cast<const ushort8*>(arow + kbase + half * 16);
        ushort8 ahi = *reinterpret_cast<const ushort8*>(arow + kbase + half * 16 + 8);
        ushort16 au = __builtin_shufflevector(alo, ahi, 0, 1, 2, 3, 4, 5, 6, 7,
                                              8, 9, 10, 11, 12, 13, 14, 15);
        v16bf a = __builtin_bit_cast(v16bf, au);

        // B fragment (32x16 bf16): lane n = lane%16; lanes 0-15 K=kbase..+15,
        // lanes 16-31 K=kbase+16..+31 -> one contiguous 32B load
        ushort16 bu = *reinterpret_cast<const ushort16*>(bcol + kbase + half * 16);
        v16bf bm = __builtin_bit_cast(v16bf, bu);

        acc = __builtin_amdgcn_wmma_f32_16x16x32_bf16(
            false, a, false, bm, (short)0, acc, false, false);
    }

    // epilogue: C layout -> M = r + 8*half, N = lane%16; fuse w_sh^gamma,
    // reduce rowsum across the 16 lanes of each half before the atomic.
#pragma unroll
    for (int r = 0; r < 8; ++r) {
        const int m = r + half * 8;
        float v = acc[r];
        float s = __powf(fmaxf(v, 0.0f), gammav[m]);
        wsharp[m * Nn + ncol] = s;
        float red = s;
        red += __shfl_xor(red, 1);
        red += __shfl_xor(red, 2);
        red += __shfl_xor(red, 4);
        red += __shfl_xor(red, 8);
        if (m16 == 0) atomicAdd(&rowsum[m], red);
    }
}

// ---------------------------------------------------------------------------
// Per-step kernel D: normalize w, update prev_w, memory erase/add, readout rv.
// One block per batch b; lane-over-d gives coalesced mem accesses.
// ---------------------------------------------------------------------------
__global__ __launch_bounds__(256) void update_kernel(float* __restrict__ mem,
                                                     const float* __restrict__ wsharp,
                                                     const float* __restrict__ rowsum,
                                                     float* __restrict__ prev_w,
                                                     const float* __restrict__ erasev,
                                                     const float* __restrict__ addv,
                                                     float* __restrict__ rv,
                                                     float* __restrict__ out,
                                                     int t) {
    const int b = blockIdx.x;
    const int tid = threadIdx.x;
    __shared__ float wl[Hh * Nn];    // 16 KB: [0..2047]=read head, [2048..]=write head
    __shared__ float rred[256];

    for (int idx = tid; idx < Hh * Nn; idx += 256) {
        int h = idx >> 11, n = idx & (Nn - 1);
        int m = b * Hh + h;
        float w = wsharp[m * Nn + n] / (rowsum[m] + 1e-8f);
        wl[idx] = w;
        prev_w[m * Nn + n] = w;
    }
    __syncthreads();

    const int d = tid & 63;
    const int nch = tid >> 6;        // 4 chunks of 512 rows
    const float er = erasev[b * Dd + d];
    const float ad = addv[b * Dd + d];
    float* memb = mem + (size_t)b * Nn * Dd;
    float racc = 0.0f;
    for (int n = nch * 512; n < (nch + 1) * 512; ++n) {
        float ww = wl[Nn + n];       // write head
        float mv = memb[(size_t)n * Dd + d];
        mv = mv * (1.0f - ww * er) + ww * ad;
        memb[(size_t)n * Dd + d] = mv;
        racc += wl[n] * mv;          // read head over updated memory
    }
    rred[tid] = racc; __syncthreads();
    if (tid < Dd) {
        float r = rred[tid] + rred[tid + 64] + rred[tid + 128] + rred[tid + 192];
        rv[b * Dd + tid] = r;
        out[((size_t)b * Tt + t) * (Uu + Dd) + Uu + tid] = r;
    }
}

// ---------------------------------------------------------------------------
extern "C" void kernel_launch(void* const* d_in, const int* in_sizes, int n_in,
                              void* d_out, int out_size, void* d_ws, size_t ws_size,
                              hipStream_t stream) {
    (void)in_sizes; (void)n_in; (void)out_size; (void)ws_size;
    const float* x      = (const float*)d_in[0];
    const float* adj    = (const float*)d_in[1];
    const float* imem   = (const float*)d_in[2];
    const float* Wx     = (const float*)d_in[3];
    const float* Wh     = (const float*)d_in[4];
    const float* bvec   = (const float*)d_in[5];
    const float* Wp     = (const float*)d_in[6];
    const float* bp     = (const float*)d_in[7];
    float* out = (float*)d_out;

    // workspace carve-up (256B aligned slabs)
    char* base = (char*)d_ws;
    size_t off = 0;
    auto take = [&](size_t bytes) { char* p = base + off; off = (off + bytes + 255) & ~(size_t)255; return p; };
    unsigned short* adjT   = (unsigned short*)take((size_t)Nn * Nn * 2);          // 8 MB bf16 norm_adj^T
    float*          mem    = (float*)take((size_t)Bx * Nn * Dd * 4);              // 4 MB
    float*          prev_w = (float*)take((size_t)Bx * Hh * Nn * 4);
    float*          wsharp = (float*)take((size_t)Bx * Hh * Nn * 4);
    unsigned short* wgA    = (unsigned short*)take((size_t)Bx * Hh * Nn * 2);
    float*          hbuf   = (float*)take((size_t)Bx * Uu * 4);
    float*          cbuf   = (float*)take((size_t)Bx * Uu * 4);
    float*          rv     = (float*)take((size_t)Bx * Dd * 4);
    float*          khat   = (float*)take((size_t)Bx * Hh * Dd * 4);
    float*          betav  = (float*)take(Bx * Hh * 4);
    float*          gvv    = (float*)take(Bx * Hh * 4);
    float*          gammav = (float*)take(Bx * Hh * 4);
    float*          erasev = (float*)take((size_t)Bx * Dd * 4);
    float*          addv   = (float*)take((size_t)Bx * Dd * 4);
    float*          rowsum = (float*)take(Bx * Hh * 4);

    // one-time (per call) prep
    prep_adj_kernel<<<Nn / 256, 256, 0, stream>>>(adj, adjT);
    init_state_kernel<<<(Bx * Nn * Dd + 255) / 256, 256, 0, stream>>>(imem, mem, prev_w, hbuf, cbuf, rv);

    for (int t = 0; t < Tt; ++t) {
        cell_kernel<<<1, 1024, 0, stream>>>(x, Wx, Wh, bvec, Wp, bp, rv,
                                            hbuf, cbuf, khat, betav, gvv, gammav,
                                            erasev, addv, out, t);
        address_kernel<<<Bx * Hh, 256, 0, stream>>>(mem, khat, betav, gvv, prev_w, wgA, rowsum);
        graph_gemm_kernel<<<Nn / 16, 32, 0, stream>>>(wgA, adjT, gammav, wsharp, rowsum);
        update_kernel<<<Bx, 256, 0, stream>>>(mem, wsharp, rowsum, prev_w, erasev, addv, rv, out, t);
    }
}
